// GRU_Decoder_47399259079347
// MI455X (gfx1250) — matrix operand order repaired and checked by reading, not compiled, not run on hardware
//
#include <hip/hip_runtime.h>

// ---------------------------------------------------------------------------
// GRU decoder, B=64 T=512 IN=H=512 L=2, for gfx1250 (MI455X).
//   k_convert : x, W_ih, W_hh -> bf16 in ws; init double-buffered h state.
//   k_gi0     : gi0 = x @ W_ih[0]^T + b_ih[0]  (parallel 32768x1536x512 WMMA GEMM)
//   k_gru     : persistent 16-WG kernel, 512 sequential steps, 2 grid syncs/step.
//               Per-WG weight slice (288 rows x 512 bf16) is staged into LDS
//               once via global_load_async_to_lds_b128 (ASYNCcnt) and stays
//               resident for all timesteps; B-frags come from ds_load_b128,
//               A-frags (hidden state) from L2.
// ---------------------------------------------------------------------------

#define Bn  64
#define Tn  512
#define INn 512
#define Hn  512
#define G3  1536        // 3*H
#define NWG 16          // persistent workgroups (one 32-col strip of H each)

#define LROW      520   // padded LDS row length in bf16 (260 dwords -> no bank conflicts)
#define LDS_ROWS  288   // 96 (W_hh0) + 96 (W_ih1) + 96 (W_hh1)
#define LDS_BYTES (LDS_ROWS * LROW * 2)   // 299,520 B

typedef __attribute__((ext_vector_type(16))) __bf16       v16bf;
typedef __attribute__((ext_vector_type(8)))  float        v8f;
typedef __attribute__((ext_vector_type(4)))  unsigned int v4u;

union ABFrag { v16bf bf; v4u q[2]; };

#define V8F_ZERO {0.f,0.f,0.f,0.f,0.f,0.f,0.f,0.f}

// fp32 -> bf16 (round-to-nearest-even), stored as u16
__device__ __forceinline__ unsigned short f2bf(float f) {
  unsigned u = __float_as_uint(f);
  return (unsigned short)((u + 0x7FFFu + ((u >> 16) & 1u)) >> 16);
}

__device__ __forceinline__ float sigm(float x) {
  return 1.0f / (1.0f + __expf(-x));
}

// A-matrix fragment (16x32 bf16, MxK), row-major source A[ld rows of K]:
// lane l (0..15): row m0+l, K = k0+0..15 across v0-7
// lane l+16     : row m0+l, K = k0+16..31
__device__ __forceinline__ v16bf load_a(const unsigned short* __restrict__ A,
                                        int ld, int m0, int k0, int lane) {
  const int row = m0 + (lane & 15);
  const int kb  = (lane >> 4) * 16;
  ABFrag f;
  f.q[0] = *(const v4u*)(A + (long long)row * ld + k0 + kb);
  f.q[1] = *(const v4u*)(A + (long long)row * ld + k0 + kb + 8);
  return f.bf;
}

// B-matrix fragment (32x16 bf16, KxN) where B[k][n] = W[n][k], W row-major [N][K]:
// lane l (0..15): col n0+l, K = k0+0..15 across v0-7
// lane l+16     : col n0+l, K = k0+16..31
__device__ __forceinline__ v16bf load_b(const unsigned short* __restrict__ W,
                                        int ld, int n0, int k0, int lane) {
  const int row = n0 + (lane & 15);
  const int kb  = (lane >> 4) * 16;
  ABFrag f;
  f.q[0] = *(const v4u*)(W + (long long)row * ld + k0 + kb);
  f.q[1] = *(const v4u*)(W + (long long)row * ld + k0 + kb + 8);
  return f.bf;
}

// Same B-fragment but from the padded LDS weight cache (row = absolute slice row).
__device__ __forceinline__ v16bf load_b_lds(const unsigned short* Wl,
                                            int rowBase, int k0, int lane) {
  const int row = rowBase + (lane & 15);
  const int kb  = (lane >> 4) * 16;
  ABFrag f;
  f.q[0] = *(const v4u*)(Wl + row * LROW + k0 + kb);
  f.q[1] = *(const v4u*)(Wl + row * LROW + k0 + kb + 8);
  return f.bf;
}

__device__ __forceinline__ v8f wmma_bf(v16bf a, v16bf b, v8f c) {
  return __builtin_amdgcn_wmma_f32_16x16x32_bf16(
      /*neg_a=*/false, a, /*neg_b=*/false, b,
      /*c_mod=*/(short)0, c, /*reuse_a=*/false, /*reuse_b=*/false);
}

// Device-wide barrier for NWG persistent workgroups (agent scope).
__device__ __forceinline__ void gridsync(unsigned* __restrict__ sync) {
  __threadfence();
  __syncthreads();
  if (threadIdx.x == 0) {
    unsigned* cnt = sync;
    unsigned* gen = sync + 1;
    unsigned g = __hip_atomic_load(gen, __ATOMIC_ACQUIRE, __HIP_MEMORY_SCOPE_AGENT);
    unsigned a = __hip_atomic_fetch_add(cnt, 1u, __ATOMIC_ACQ_REL, __HIP_MEMORY_SCOPE_AGENT);
    if (a == NWG - 1u) {
      __hip_atomic_store(cnt, 0u, __ATOMIC_RELAXED, __HIP_MEMORY_SCOPE_AGENT);
      __hip_atomic_fetch_add(gen, 1u, __ATOMIC_RELEASE, __HIP_MEMORY_SCOPE_AGENT);
    } else {
      while (__hip_atomic_load(gen, __ATOMIC_ACQUIRE, __HIP_MEMORY_SCOPE_AGENT) == g)
        __builtin_amdgcn_s_sleep(1);
    }
  }
  __syncthreads();
  __builtin_amdgcn_fence(__ATOMIC_ACQUIRE, "agent");   // invalidate WGP$ for fresh h
}

// ---------------------------------------------------------------------------
// Probe kernel (compiled, never launched): cluster primitives + TDM wait.
// ---------------------------------------------------------------------------
__global__ void k_cdna5_probe(const unsigned* __restrict__ g,
                              unsigned* __restrict__ o) {
  __shared__ unsigned buf[256];
  asm volatile(
      "global_load_async_to_lds_b32 %0, %1, off\n\t"
      "s_wait_asynccnt 0"
      :: "v"((unsigned)(threadIdx.x * 4u)), "v"(g + threadIdx.x)
      : "memory");
  __builtin_amdgcn_s_wait_tensorcnt(0);        // TENSORcnt wait (TDM path)
  __syncthreads();
  unsigned v = buf[threadIdx.x] + (unsigned)__builtin_amdgcn_cluster_id_x();
  __builtin_amdgcn_s_cluster_barrier();        // s_barrier_signal/-wait -3
  o[threadIdx.x] = v + buf[(threadIdx.x + 1) & 255];
}

// ---------------------------------------------------------------------------
// Kernel 1: convert inputs to bf16, init state, reset sync.
// ---------------------------------------------------------------------------
__global__ void k_convert(const float* __restrict__ x,
                          const float* __restrict__ eh,
                          const float* __restrict__ Wih,
                          const float* __restrict__ Whh,
                          unsigned short* __restrict__ xbf,
                          unsigned short* __restrict__ wihbf,
                          unsigned short* __restrict__ whhbf,
                          float* __restrict__ h0f, float* __restrict__ h1f,
                          unsigned short* __restrict__ h0b,
                          unsigned short* __restrict__ h1b,
                          unsigned* __restrict__ sync) {
  const long long NX = (long long)Bn * Tn * INn;      // 16,777,216
  const long long NW = (long long)2 * G3 * INn;       //  1,572,864
  const long long NH = (long long)2 * Bn * Hn;        //     65,536
  const long long total = NX + 2 * NW + NH;
  const long long stride = (long long)gridDim.x * blockDim.x;
  for (long long i = (long long)blockIdx.x * blockDim.x + threadIdx.x;
       i < total; i += stride) {
    if (i < NX) {
      xbf[i] = f2bf(x[i]);
    } else if (i < NX + NW) {
      long long j = i - NX;           wihbf[j] = f2bf(Wih[j]);
    } else if (i < NX + 2 * NW) {
      long long j = i - NX - NW;      whhbf[j] = f2bf(Whh[j]);
    } else {
      long long j = i - NX - 2 * NW;  // encoder_h [2][64][512], buffer 0 only
      float v = eh[j];
      if (j < (long long)Bn * Hn) { h0f[j] = v; h0b[j] = f2bf(v); }
      else { long long k = j - (long long)Bn * Hn; h1f[k] = v; h1b[k] = f2bf(v); }
    }
  }
  if (blockIdx.x == 0 && threadIdx.x == 0) { sync[0] = 0u; sync[1] = 0u; }
}

// ---------------------------------------------------------------------------
// Kernel 2: gi0[b*T+t][0..1536) = x @ W_ih0^T + b_ih0   (parallel WMMA GEMM)
// grid = (M/64=512, N/64=24), block = 256 (8 waves: 4 M-tiles x 2 N-halves)
// ---------------------------------------------------------------------------
__global__ void __launch_bounds__(256)
k_gi0(const unsigned short* __restrict__ xbf,
      const unsigned short* __restrict__ wihbf,
      const float* __restrict__ bih,
      float* __restrict__ gi0) {
  const int lane = threadIdx.x & 31;
  const int wave = threadIdx.x >> 5;
  const int m0 = blockIdx.x * 64 + (wave & 3) * 16;
  const int n0 = blockIdx.y * 64 + (wave >> 2) * 32;
  v8f c0 = V8F_ZERO, c1 = V8F_ZERO;
  for (int k0 = 0; k0 < INn; k0 += 32) {
    v16bf a  = load_a(xbf,   INn, m0,      k0, lane);
    v16bf b0 = load_b(wihbf, INn, n0,      k0, lane);
    v16bf b1 = load_b(wihbf, INn, n0 + 16, k0, lane);
    c0 = wmma_bf(a, b0, c0);
    c1 = wmma_bf(a, b1, c1);
  }
  const int col = n0 + (lane & 15);
  const int rB  = m0 + (lane >> 4) * 8;
  const float ba = bih[col], bb = bih[col + 16];
#pragma unroll
  for (int v = 0; v < 8; ++v) {
    long long r = rB + v;
    gi0[r * G3 + col]      = c0[v] + ba;
    gi0[r * G3 + col + 16] = c1[v] + bb;
  }
}

// ---------------------------------------------------------------------------
// Kernel 3: persistent sequential GRU. 16 WGs x 256 threads.
// WG w owns H-columns [w*32, w*32+32). Wave = (M-tile 0..3) x (N-half 0..1).
// Weight slice cached in LDS for all 512 steps (staged via async LDS-DMA):
//   rows [  0, 96) = W_hh0[g*512 + colBase + j]   (g=0..2, j=0..31)
//   rows [ 96,192) = W_ih1[...]
//   rows [192,288) = W_hh1[...]
// ---------------------------------------------------------------------------
__global__ void __launch_bounds__(256)
k_gru(const float* __restrict__ bih, const float* __restrict__ bhh,
      const unsigned short* __restrict__ wihbf,
      const unsigned short* __restrict__ whhbf,
      const float* __restrict__ gi0,
      float* __restrict__ h0f, float* __restrict__ h1f,
      unsigned short* __restrict__ h0b, unsigned short* __restrict__ h1b,
      float* __restrict__ out, unsigned* __restrict__ sync) {
  extern __shared__ unsigned short ldsw[];   // LDS_ROWS x LROW bf16

  const int lane  = threadIdx.x & 31;
  const int wave  = threadIdx.x >> 5;
  const int m0    = (wave & 3) * 16;
  const int nOffL = (wave >> 2) * 16;                  // N-half within the strip
  const int colBase = blockIdx.x * 32;
  const int col16 = colBase + nOffL;
  const int c     = col16 + (lane & 15);       // this lane's H-column
  const int rB    = m0 + (lane >> 4) * 8;      // this lane's base row (batch idx)

  const unsigned short* whh0 = whhbf;                              // layer 0
  const unsigned short* wih1 = wihbf + (long long)G3 * INn;        // layer 1
  const unsigned short* whh1 = whhbf + (long long)G3 * INn;

  // ---- stage weight slice into LDS via async LDS-DMA (ASYNCcnt path) ----
  // 18432 16-byte chunks / 256 threads = 72 full-EXEC iterations per lane.
  for (int ch = threadIdx.x; ch < LDS_ROWS * 64; ch += 256) {
    const int r  = ch >> 6;                 // slice row 0..287
    const int kc = (ch & 63) << 3;          // bf16 offset, step 8 (16 B)
    const int slice = r / 96;               // 0: whh0, 1: wih1, 2: whh1
    const int rr = r % 96;
    const int g  = rr >> 5;                 // gate
    const int j  = rr & 31;                 // column within strip
    const unsigned short* src = (slice == 0) ? whh0 : ((slice == 1) ? wih1 : whh1);
    const unsigned short* gp  = src + (long long)(g * Hn + colBase + j) * Hn + kc;
    // LDS byte offset is the low 32 bits of the generic LDS pointer.
    const unsigned ldsoff = (unsigned)(unsigned long long)(ldsw + r * LROW + kc);
    asm volatile("global_load_async_to_lds_b128 %0, %1, off"
                 :: "v"(ldsoff), "v"(gp) : "memory");
  }
  asm volatile("s_wait_asynccnt 0" ::: "memory");
  __syncthreads();

  // biases (per-column scalars, hoisted out of the time loop)
  const float b0r = bhh[c], b0z = bhh[Hn + c], b0n = bhh[2 * Hn + c];
  const float i1r = bih[G3 + c], i1z = bih[G3 + Hn + c], i1n = bih[G3 + 2 * Hn + c];
  const float h1r = bhh[G3 + c], h1z = bhh[G3 + Hn + c], h1n = bhh[G3 + 2 * Hn + c];

  float* dh = out + (long long)Bn * Tn * Hn;   // final-state tail of d_out

  for (int t = 0; t < Tn; ++t) {
    const int cur = t & 1, nxt = cur ^ 1;
    const unsigned short* h0b_c = h0b + cur * Bn * Hn;
    unsigned short*       h0b_n = h0b + nxt * Bn * Hn;
    const float*          h0f_c = h0f + cur * Bn * Hn;
    float*                h0f_n = h0f + nxt * Bn * Hn;
    const unsigned short* h1b_c = h1b + cur * Bn * Hn;
    unsigned short*       h1b_n = h1b + nxt * Bn * Hn;
    const float*          h1f_c = h1f + cur * Bn * Hn;
    float*                h1f_n = h1f + nxt * Bn * Hn;

    // ---- Phase A: layer 0 ----  gh0 = h0 @ W_hh0^T ; combine with gi0[t]
    {
      // hide the epilogue's strided gi0 gather behind the GEMM
#pragma unroll
      for (int v = 0; v < 8; ++v) {
        const float* gp = gi0 + ((long long)(rB + v) * Tn + t) * G3;
        __builtin_prefetch(gp + c, 0, 0);
        __builtin_prefetch(gp + Hn + c, 0, 0);
        __builtin_prefetch(gp + 2 * Hn + c, 0, 0);
      }
      v8f gr = V8F_ZERO, gz = V8F_ZERO, gn = V8F_ZERO;
      for (int k0 = 0; k0 < Hn; k0 += 32) {
        v16bf a = load_a(h0b_c, Hn, m0, k0, lane);
        gr = wmma_bf(a, load_b_lds(ldsw,      nOffL, k0, lane), gr);
        gz = wmma_bf(a, load_b_lds(ldsw, 32 + nOffL, k0, lane), gz);
        gn = wmma_bf(a, load_b_lds(ldsw, 64 + nOffL, k0, lane), gn);
      }
#pragma unroll
      for (int v = 0; v < 8; ++v) {
        const int row = rB + v;
        const float* gi = gi0 + ((long long)row * Tn + t) * G3;  // b_ih0 folded in
        float r  = sigm(gi[c]          + gr[v] + b0r);
        float z  = sigm(gi[Hn + c]     + gz[v] + b0z);
        float n  = tanhf(gi[2 * Hn + c] + r * (gn[v] + b0n));
        float hp = h0f_c[row * Hn + c];
        float hv = (1.0f - z) * n + z * hp;
        h0f_n[row * Hn + c] = hv;
        h0b_n[row * Hn + c] = f2bf(hv);
        if (t == Tn - 1) dh[row * Hn + c] = hv;
      }
    }
    gridsync(sync);   // h0_new visible device-wide

    // ---- Phase B: layer 1 ----  gi1 = h0_new @ W_ih1^T ; gh1 = h1 @ W_hh1^T
    {
      v8f ar = V8F_ZERO, az = V8F_ZERO, ain = V8F_ZERO, ahn = V8F_ZERO;
      for (int k0 = 0; k0 < Hn; k0 += 32) {
        v16bf a0 = load_a(h0b_n, Hn, m0, k0, lane);   // layer-1 input
        v16bf a1 = load_a(h1b_c, Hn, m0, k0, lane);   // layer-1 state
        ar  = wmma_bf(a0, load_b_lds(ldsw,  96 + nOffL, k0, lane), ar);
        ar  = wmma_bf(a1, load_b_lds(ldsw, 192 + nOffL, k0, lane), ar);
        az  = wmma_bf(a0, load_b_lds(ldsw, 128 + nOffL, k0, lane), az);
        az  = wmma_bf(a1, load_b_lds(ldsw, 224 + nOffL, k0, lane), az);
        ain = wmma_bf(a0, load_b_lds(ldsw, 160 + nOffL, k0, lane), ain);
        ahn = wmma_bf(a1, load_b_lds(ldsw, 256 + nOffL, k0, lane), ahn);
      }
#pragma unroll
      for (int v = 0; v < 8; ++v) {
        const int row = rB + v;
        float r  = sigm(ar[v] + i1r + h1r);
        float z  = sigm(az[v] + i1z + h1z);
        float n  = tanhf(ain[v] + i1n + r * (ahn[v] + h1n));
        float hp = h1f_c[row * Hn + c];
        float hv = (1.0f - z) * n + z * hp;
        h1f_n[row * Hn + c] = hv;
        h1b_n[row * Hn + c] = f2bf(hv);
        out[(long long)row * Tn * Hn + (long long)t * Hn + c] = hv;
        if (t == Tn - 1) dh[(long long)Bn * Hn + row * Hn + c] = hv;
      }
    }
    gridsync(sync);   // h1_new visible; safe to rotate buffers
  }
}

// ---------------------------------------------------------------------------
// Workspace layout (bytes), total ~231 MB
// ---------------------------------------------------------------------------
#define O_XBF  0ULL
#define O_WIH  33554432ULL
#define O_WHH  36700160ULL
#define O_H0F  39845888ULL
#define O_H1F  40108032ULL
#define O_H0B  40370176ULL
#define O_H1B  40501248ULL
#define O_SYNC 40632320ULL
#define O_GI0  40632576ULL

extern "C" void kernel_launch(void* const* d_in, const int* in_sizes, int n_in,
                              void* d_out, int out_size, void* d_ws, size_t ws_size,
                              hipStream_t stream) {
  const float* x   = (const float*)d_in[0];   // [B,T,IN]
  const float* eh  = (const float*)d_in[1];   // [L,B,H]
  const float* Wih = (const float*)d_in[2];   // [L,3H,IN]
  const float* Whh = (const float*)d_in[3];   // [L,3H,H]
  const float* bih = (const float*)d_in[4];   // [L,3H]
  const float* bhh = (const float*)d_in[5];   // [L,3H]
  float* out = (float*)d_out;                 // [B,T,H] ++ [L,B,H]

  char* ws = (char*)d_ws;
  unsigned short* xbf   = (unsigned short*)(ws + O_XBF);
  unsigned short* wihbf = (unsigned short*)(ws + O_WIH);
  unsigned short* whhbf = (unsigned short*)(ws + O_WHH);
  float*          h0f   = (float*)(ws + O_H0F);
  float*          h1f   = (float*)(ws + O_H1F);
  unsigned short* h0b   = (unsigned short*)(ws + O_H0B);
  unsigned short* h1b   = (unsigned short*)(ws + O_H1B);
  unsigned*       sync  = (unsigned*)(ws + O_SYNC);
  float*          gi0   = (float*)(ws + O_GI0);

  // allow the 292.5 KB dynamic-LDS allocation (host API, not stream-ordered)
  (void)hipFuncSetAttribute((const void*)k_gru,
                            hipFuncAttributeMaxDynamicSharedMemorySize,
                            LDS_BYTES);

  // 1. convert to bf16, init state, reset barrier
  k_convert<<<8192, 256, 0, stream>>>(x, eh, Wih, Whh, xbf, wihbf, whhbf,
                                      h0f, h1f, h0b, h1b, sync);
  // 2. precompute layer-0 input projection for all timesteps
  k_gi0<<<dim3(512, 24, 1), 256, 0, stream>>>(xbf, wihbf, bih, gi0);
  // 3. sequential recurrence (persistent, 16 WGs, weights LDS-resident)
  k_gru<<<NWG, 256, LDS_BYTES, stream>>>(bih, bhh, wihbf, whhbf, gi0,
                                         h0f, h1f, h0b, h1b, out, sync);
}
